// _DomainResidualAdapterBase_9972914061663
// MI455X (gfx1250) — matrix operand order repaired and checked by reading, not compiled, not run on hardware
//
#include <hip/hip_runtime.h>
#include <hip/hip_bf16.h>

// Problem constants (from reference): D,K,F,H,N,C = 8,16,768,768,16384,128
constexpr int ND = 8;      // domains
constexpr int NK = 16;     // cols per domain
constexpr int NF = 768;    // feature dim
constexpr int NH = 768;    // hidden dim
constexpr int NN = 16384;  // rows
constexpr int NC = 128;    // channels
constexpr int KP = 800;    // 771 (F+3) padded to 25*32 for bf16 WMMA K-chunks

// LDS row strides (shorts). 8*odd shorts = 16B-aligned AND 4*odd dwords, so the
// 16 row-starts of a ds_load_b128 fragment cover all 64 LDS banks exactly once.
constexpr int SASTR = 808; // x tile stride (>= 800)
constexpr int SHSTR = 776; // h tile stride (>= 768)

typedef __attribute__((ext_vector_type(16))) __bf16 v16bf;
typedef __attribute__((ext_vector_type(8)))  float  v8f;

union Frag { v16bf v; uint4 q[2]; };

__device__ __forceinline__ unsigned short f2bf(float f) {
  unsigned int u = __float_as_uint(f);
  unsigned int r = (u + 0x7FFFu + ((u >> 16) & 1u)) >> 16;
  return (unsigned short)r;
}

// ---------------------------------------------------------------------------
// Kernel 1: zero bucket counters, copy z_base -> z_out, convert feats -> bf16
// into padded x buffer (columns 0..767).
// ---------------------------------------------------------------------------
__global__ void k_prep(const float* __restrict__ zb, const float* __restrict__ feats,
                       float* __restrict__ zout, unsigned short* __restrict__ xb,
                       int* __restrict__ counters) {
  int tid = blockIdx.x * blockDim.x + threadIdx.x;
  int stride = gridDim.x * blockDim.x;
  if (tid < ND) counters[tid] = 0;

  int total4 = NN * NC / 4;
  for (int i = tid; i < total4; i += stride)
    ((float4*)zout)[i] = ((const float4*)zb)[i];

  int totalf = NN * (NF / 8);
  for (int i = tid; i < totalf; i += stride) {
    int row = i / (NF / 8);
    int c8  = i % (NF / 8);
    const float4* src = (const float4*)(feats + (size_t)row * NF + c8 * 8);
    float4 a = src[0], b = src[1];
    uint4 o;
    o.x = (unsigned int)f2bf(a.x) | ((unsigned int)f2bf(a.y) << 16);
    o.y = (unsigned int)f2bf(a.z) | ((unsigned int)f2bf(a.w) << 16);
    o.z = (unsigned int)f2bf(b.x) | ((unsigned int)f2bf(b.y) << 16);
    o.w = (unsigned int)f2bf(b.z) | ((unsigned int)f2bf(b.w) << 16);
    *(uint4*)(xb + (size_t)row * KP + c8 * 8) = o;
  }
}

// ---------------------------------------------------------------------------
// Kernel 2a: tiled transpose W1 (D,771,768) f32 -> W1bT (D,768,800) bf16.
// 32x32 tiles through LDS: both global read and write sides coalesced.
// grid (KP/32, NH/32, D), block (32,8).
// ---------------------------------------------------------------------------
__global__ void k_w1t(const float* __restrict__ W1, unsigned short* __restrict__ W1bT) {
  __shared__ float t[32][33];
  const int d  = blockIdx.z;
  const int k0 = blockIdx.x * 32;
  const int h0 = blockIdx.y * 32;
  for (int i = threadIdx.y; i < 32; i += 8) {
    int k = k0 + i;
    int h = h0 + threadIdx.x;                     // consecutive -> coalesced
    t[i][threadIdx.x] = (k < NF + 3) ? W1[((size_t)d * (NF + 3) + k) * NH + h] : 0.0f;
  }
  __syncthreads();
  for (int i = threadIdx.y; i < 32; i += 8) {
    int h = h0 + i;
    int k = k0 + threadIdx.x;                     // consecutive -> coalesced
    W1bT[((size_t)d * NH + h) * KP + k] = f2bf(t[threadIdx.x][i]);
  }
}

// ---------------------------------------------------------------------------
// Kernel 2b: W2 (D,768,16) -> W2bT (D,16,768) bf16 (tiny: 98K elements).
// ---------------------------------------------------------------------------
__global__ void k_w2t(const float* __restrict__ W2, unsigned short* __restrict__ W2bT) {
  int tid = blockIdx.x * blockDim.x + threadIdx.x;
  int stride = gridDim.x * blockDim.x;
  int total = ND * NK * NH;
  for (int o = tid; o < total; o += stride) {
    int d = o / (NK * NH);
    int rem = o % (NK * NH);
    int n = rem / NH;
    int k = rem % NH;
    W2bT[o] = f2bf(W2[((size_t)d * NH + k) * NK + n]);
  }
}

// ---------------------------------------------------------------------------
// Kernel 3: per-row softmax confidence features + bucket rows by domain.
// ---------------------------------------------------------------------------
__global__ void k_conf(const float* __restrict__ zb, const int* __restrict__ dom,
                       unsigned short* __restrict__ xb, int* __restrict__ counters,
                       int* __restrict__ idx) {
  int row = blockIdx.x * blockDim.x + threadIdx.x;
  if (row >= NN) return;
  int d = dom[row];
  const float* p = zb + (size_t)row * NC + d * NK;
  float l[NK];
  float m = -3.0e38f;
  #pragma unroll
  for (int i = 0; i < NK; ++i) { l[i] = p[i]; m = fmaxf(m, l[i]); }
  float s = 0.f;
  #pragma unroll
  for (int i = 0; i < NK; ++i) { l[i] -= m; s += __expf(l[i]); }
  float inv = 1.0f / s;
  float logs = __logf(s);
  float t1 = -3.0e38f, t2 = -3.0e38f, ent = 0.f;
  #pragma unroll
  for (int i = 0; i < NK; ++i) {
    float pi = __expf(l[i]) * inv;
    ent -= pi * (l[i] - logs);
    if (pi > t1) { t2 = t1; t1 = pi; }
    else if (pi > t2) { t2 = pi; }
  }
  ent *= 0.36067376022224085f;         // 1/log(16)
  unsigned short* xr = xb + (size_t)row * KP;
  xr[NF + 0] = f2bf(t1);
  xr[NF + 1] = f2bf(t1 - t2);
  xr[NF + 2] = f2bf(ent);
  #pragma unroll
  for (int i = NF + 3; i < KP; ++i) xr[i] = 0;
  int slot = atomicAdd(&counters[d], 1);
  idx[d * NN + slot] = row;
}

// ---------------------------------------------------------------------------
// Kernel 4: per (domain, 32-row tile) fused WMMA GEMM.
//   h = relu(x @ W1[d] + b1[d]); dz = (h @ W2[d] + b2[d]) * alpha[d]
// 8 waves; wave w owns N-tiles 6w..6w+5 across BOTH 16-row M-subtiles
// (12 WMMAs per K-chunk per wave, B fragments batch-loaded up front).
// h overlays the x LDS buffer (x dead after GEMM1). Split-K GEMM2 reduced
// with ds_add_f32 atomics.
// ---------------------------------------------------------------------------
__launch_bounds__(256, 1)
__global__ void k_gemm(const float* __restrict__ zb, const unsigned short* __restrict__ xb,
                       const unsigned short* __restrict__ W1bT, const unsigned short* __restrict__ W2bT,
                       const float* __restrict__ b1, const float* __restrict__ b2,
                       const float* __restrict__ alphas, const int* __restrict__ counters,
                       const int* __restrict__ idx, float* __restrict__ zout) {
  __shared__ unsigned short sBuf[32 * SASTR];  // 51712 B: x tile, then h tile overlay
  __shared__ float sDZ[2 * 256];               // dz accumulators (2 M-subtiles)
  __shared__ int sR[32];

  const int d = blockIdx.y;
  const int count = counters[d];
  const int row0 = blockIdx.x * 32;
  if (row0 >= count) return;                   // uniform per block

  const int tid = threadIdx.x;
  if (tid < 32) {
    int rr = row0 + tid;
    sR[tid] = idx[d * NN + (rr < count ? rr : row0)];
  }
  for (int i = tid; i < 512; i += 256) sDZ[i] = 0.0f;
  __syncthreads();

  // Stage the 32 x 800 bf16 activation tile into LDS (b128 on both sides).
  for (int i = tid; i < 32 * (KP / 8); i += 256) {
    int m = i / (KP / 8);
    int c = i % (KP / 8);
    uint4 v = *(const uint4*)(xb + (size_t)sR[m] * KP + c * 8);
    *(uint4*)(&sBuf[m * SASTR + c * 8]) = v;
  }
  __syncthreads();

  const int wave = tid >> 5;
  const int lane = tid & 31;
  const int mrow = lane & 15;                  // M for A-frag, N for B-frag
  const int hi   = lane >> 4;

  v8f zero = {};
  v8f acc0[6], acc1[6];
  #pragma unroll
  for (int j = 0; j < 6; ++j) { acc0[j] = zero; acc1[j] = zero; }

  // Precompute B base pointers once (kills per-iteration 64-bit MADs).
  const unsigned short* bp1[6];
  #pragma unroll
  for (int j = 0; j < 6; ++j) {
    int n = (wave * 6 + j) * 16 + mrow;        // hidden column for this lane
    bp1[j] = W1bT + ((size_t)d * NH + n) * KP + hi * 8;
  }

  // GEMM1: h(32x768) = x(32x800) @ W1(800x768), K = 25 chunks of 32.
  for (int kc = 0; kc < KP / 32; ++kc) {
    // Batch all 12 global b128 loads for this K-chunk before any WMMA.
    Frag b[6];
    #pragma unroll
    for (int j = 0; j < 6; ++j) {
      const uint4* q = (const uint4*)(bp1[j] + kc * 32);
      b[j].q[0] = q[0];
      b[j].q[1] = q[2];
    }
    Frag a0, a1;
    int ka = kc * 32 + hi * 8;
    a0.q[0] = *(const uint4*)(&sBuf[mrow * SASTR + ka]);
    a0.q[1] = *(const uint4*)(&sBuf[mrow * SASTR + ka + 16]);
    a1.q[0] = *(const uint4*)(&sBuf[(mrow + 16) * SASTR + ka]);
    a1.q[1] = *(const uint4*)(&sBuf[(mrow + 16) * SASTR + ka + 16]);
    #pragma unroll
    for (int j = 0; j < 6; ++j) {
      acc0[j] = __builtin_amdgcn_wmma_f32_16x16x32_bf16(
          false, a0.v, false, b[j].v, (short)0, acc0[j], false, false);
      acc1[j] = __builtin_amdgcn_wmma_f32_16x16x32_bf16(
          false, a1.v, false, b[j].v, (short)0, acc1[j], false, false);
    }
  }

  __syncthreads();   // everyone done reading x tile -> safe to overlay h

  // Epilogue 1: +b1, relu, bf16, store into the h overlay.
  #pragma unroll
  for (int j = 0; j < 6; ++j) {
    int ncol = (wave * 6 + j) * 16 + mrow;
    float bb = b1[d * NH + ncol];
    int mbase = hi * 8;
    #pragma unroll
    for (int v = 0; v < 8; ++v) {
      float h0 = fmaxf(acc0[j][v] + bb, 0.0f);
      float h1 = fmaxf(acc1[j][v] + bb, 0.0f);
      sBuf[(mbase + v) * SHSTR + ncol]        = f2bf(h0);
      sBuf[(16 + mbase + v) * SHSTR + ncol]   = f2bf(h1);
    }
  }
  __syncthreads();

  // GEMM2 (split-K): dz(32x16) = h(32x768) @ W2(768x16); wave w: chunks 3w..3w+2.
  v8f accD0 = zero, accD1 = zero;
  const unsigned short* bp2 = W2bT + ((size_t)d * NK + mrow) * NH + hi * 8;
  #pragma unroll
  for (int kc = 0; kc < 3; ++kc) {
    int kk = (wave * 3 + kc) * 32;
    Frag b;
    const uint4* q = (const uint4*)(bp2 + kk);
    b.q[0] = q[0];
    b.q[1] = q[2];
    Frag a0, a1;
    int ka = kk + hi * 8;
    a0.q[0] = *(const uint4*)(&sBuf[mrow * SHSTR + ka]);
    a0.q[1] = *(const uint4*)(&sBuf[mrow * SHSTR + ka + 16]);
    a1.q[0] = *(const uint4*)(&sBuf[(mrow + 16) * SHSTR + ka]);
    a1.q[1] = *(const uint4*)(&sBuf[(mrow + 16) * SHSTR + ka + 16]);
    accD0 = __builtin_amdgcn_wmma_f32_16x16x32_bf16(
        false, a0.v, false, b.v, (short)0, accD0, false, false);
    accD1 = __builtin_amdgcn_wmma_f32_16x16x32_bf16(
        false, a1.v, false, b.v, (short)0, accD1, false, false);
  }
  {
    int mbase = hi * 8;
    #pragma unroll
    for (int v = 0; v < 8; ++v) {
      atomicAdd(&sDZ[(mbase + v) * 16 + mrow],       accD0[v]);
      atomicAdd(&sDZ[256 + (mbase + v) * 16 + mrow], accD1[v]);
    }
  }
  __syncthreads();

  // Scale + scatter into the row's 16 domain columns.
  const int Ml = tid >> 4, Nn = tid & 15;      // 256 threads == 16x16
  const float bia = b2[d * NK + Nn];
  const float alp = alphas[d];
  #pragma unroll
  for (int mt = 0; mt < 2; ++mt) {
    int rloc = mt * 16 + Ml;
    if (row0 + rloc < count) {
      float dz = (sDZ[mt * 256 + Ml * 16 + Nn] + bia) * alp;
      int r = sR[rloc];
      int col = d * NK + Nn;
      zout[(size_t)r * NC + col] = zb[(size_t)r * NC + col] + dz;
    }
  }
}

// ---------------------------------------------------------------------------
extern "C" void kernel_launch(void* const* d_in, const int* in_sizes, int n_in,
                              void* d_out, int out_size, void* d_ws, size_t ws_size,
                              hipStream_t stream) {
  const float* zb     = (const float*)d_in[0];   // z_base_global (N,C) f32
  const int*   dom    = (const int*)  d_in[1];   // domain_ids (N,) i32
  const float* feats  = (const float*)d_in[2];   // feats (N,F) f32
  // d_in[3] = cols (deterministic arange, not needed)
  const float* W1     = (const float*)d_in[4];   // (D, F+3, H) f32
  const float* b1     = (const float*)d_in[5];   // (D, H) f32
  const float* W2     = (const float*)d_in[6];   // (D, H, K) f32
  const float* b2     = (const float*)d_in[7];   // (D, K) f32
  const float* alphas = (const float*)d_in[8];   // (D,) f32
  float* zout = (float*)d_out;

  char* ws = (char*)d_ws;
  int* counters = (int*)ws;                                   // 8 ints
  int* idx      = (int*)(ws + 256);                           // D*N ints = 512 KB
  unsigned short* xb   = (unsigned short*)(ws + 256 + (size_t)ND * NN * 4);        // N*800 bf16
  unsigned short* W1bT = (unsigned short*)((char*)xb + (size_t)NN * KP * 2);       // D*768*800 bf16
  unsigned short* W2bT = (unsigned short*)((char*)W1bT + (size_t)ND * NH * KP * 2);// D*16*768 bf16

  k_prep<<<1024, 256, 0, stream>>>(zb, feats, zout, xb, counters);
  k_w1t <<<dim3(KP / 32, NH / 32, ND), dim3(32, 8), 0, stream>>>(W1, W1bT);
  k_w2t <<<96, 256, 0, stream>>>(W2, W2bT);
  k_conf<<<NN / 256, 256, 0, stream>>>(zb, dom, xb, counters, idx);
  k_gemm<<<dim3(NN / 32, ND), 256, 0, stream>>>(zb, xb, W1bT, W2bT,
                                                b1, b2, alphas, counters, idx, zout);
}
